// QATGPT2MLP_28286654611543
// MI455X (gfx1250) — 8-bit matrix operand order repaired and checked by reading, compiled, not run on hardware
//
#include <hip/hip_runtime.h>
#include <cstdint>
#include <cstddef>


typedef __attribute__((ext_vector_type(8))) int v8i;

#define M_DIM   8192
#define K1_DIM  1024
#define N1_DIM  4096
#define K2_DIM  4096
#define N2_DIM  1024

#define OFF_SCALES 0ul
#define OFF_XQ     256ul
#define OFF_WFCQ   (OFF_XQ   + 8388608ul)
#define OFF_WPQ    (OFF_WFCQ + 4194304ul)
#define OFF_H      (OFF_WPQ  + 4194304ul)
#define OFF_HQ     (OFF_H    + 134217728ul)
#define OFF_T1     (OFF_HQ   + 33554432ul)
#define OFF_T2     (OFF_T1   + 262144ul)

__device__ __forceinline__ void async_copy_b128(unsigned ldsOff, const int8_t* g) {
    asm volatile("global_load_async_to_lds_b128 %0, %1, off"
                 :: "v"(ldsOff), "v"((unsigned long long)(uintptr_t)g)
                 : "memory");
}
__device__ __forceinline__ void wait_async0() {
    asm volatile("s_wait_asynccnt 0x0" ::: "memory");
}

__global__ void qatmlp_init_kernel(unsigned* slots) {
    if (threadIdx.x < 4) slots[threadIdx.x] = 0u;
}

__global__ __launch_bounds__(256) void qatmlp_absmax_kernel(
    const float* __restrict__ x, size_t n, unsigned* __restrict__ slot) {
    __shared__ unsigned smem[256];
    unsigned local = 0u;
    size_t stride = (size_t)gridDim.x * blockDim.x;
    for (size_t i = (size_t)blockIdx.x * blockDim.x + threadIdx.x; i < n; i += stride) {
        unsigned b = __float_as_uint(x[i]) & 0x7fffffffu;
        local = local > b ? local : b;
    }
    smem[threadIdx.x] = local;
    __syncthreads();
    for (int s = 128; s > 0; s >>= 1) {
        if (threadIdx.x < (unsigned)s) {
            unsigned o = smem[threadIdx.x + s];
            if (o > smem[threadIdx.x]) smem[threadIdx.x] = o;
        }
        __syncthreads();
    }
    if (threadIdx.x == 0) atomicMax(slot, smem[0]);
}

__global__ __launch_bounds__(256) void qatmlp_quant_kernel(
    const float* __restrict__ x, int8_t* __restrict__ q, size_t n4,
    const unsigned* __restrict__ slot) {
    float amax  = __uint_as_float(*slot);
    float scale = fmaxf(amax / 127.0f, 1e-8f);
    size_t stride = (size_t)gridDim.x * blockDim.x;
    const float4* xv = (const float4*)x;
    int* qv = (int*)q;
    for (size_t i = (size_t)blockIdx.x * blockDim.x + threadIdx.x; i < n4; i += stride) {
        float4 v = xv[i];
        int q0 = (int)rintf(v.x / scale); q0 = q0 < -128 ? -128 : (q0 > 127 ? 127 : q0);
        int q1 = (int)rintf(v.y / scale); q1 = q1 < -128 ? -128 : (q1 > 127 ? 127 : q1);
        int q2 = (int)rintf(v.z / scale); q2 = q2 < -128 ? -128 : (q2 > 127 ? 127 : q2);
        int q3 = (int)rintf(v.w / scale); q3 = q3 < -128 ? -128 : (q3 > 127 ? 127 : q3);
        qv[i] = (q0 & 255) | ((q1 & 255) << 8) | ((q2 & 255) << 16) | ((q3 & 255) << 24);
    }
}

__global__ __launch_bounds__(256) void qatmlp_rowquant_lora_kernel(
    const float* __restrict__ X, const float* __restrict__ A,
    int8_t* __restrict__ Q, float* __restrict__ T, int K,
    const unsigned* __restrict__ slot) {
    __shared__ float red[8][256];
    float amax  = __uint_as_float(*slot);
    float scale = fmaxf(amax / 127.0f, 1e-8f);

    int m = blockIdx.x;
    const float4* xr = (const float4*)(X + (size_t)m * K);
    const float4* A4 = (const float4*)A;
    int* qv = (int*)(Q + (size_t)m * K);
    int K4 = K >> 2;

    float acc[8];
#pragma unroll
    for (int r = 0; r < 8; ++r) acc[r] = 0.0f;

    for (int i = threadIdx.x; i < K4; i += 256) {
        float4 v = xr[i];
#pragma unroll
        for (int r = 0; r < 8; ++r) {
            float4 a = A4[r * K4 + i];
            acc[r] += v.x * a.x + v.y * a.y + v.z * a.z + v.w * a.w;
        }
        int q0 = (int)rintf(v.x / scale); q0 = q0 < -128 ? -128 : (q0 > 127 ? 127 : q0);
        int q1 = (int)rintf(v.y / scale); q1 = q1 < -128 ? -128 : (q1 > 127 ? 127 : q1);
        int q2 = (int)rintf(v.z / scale); q2 = q2 < -128 ? -128 : (q2 > 127 ? 127 : q2);
        int q3 = (int)rintf(v.w / scale); q3 = q3 < -128 ? -128 : (q3 > 127 ? 127 : q3);
        qv[i] = (q0 & 255) | ((q1 & 255) << 8) | ((q2 & 255) << 16) | ((q3 & 255) << 24);
    }
#pragma unroll
    for (int r = 0; r < 8; ++r) red[r][threadIdx.x] = acc[r];
    __syncthreads();
    for (int s = 128; s > 0; s >>= 1) {
        if (threadIdx.x < (unsigned)s)
#pragma unroll
            for (int r = 0; r < 8; ++r) red[r][threadIdx.x] += red[r][threadIdx.x + s];
        __syncthreads();
    }
    if (threadIdx.x < 8) T[(size_t)m * 8 + threadIdx.x] = red[threadIdx.x][0];
}

#define KSLAB  128
#define TPITCH 144

template <bool FUSE_GELU_ABSMAX>
__global__ __launch_bounds__(256) void qatmlp_gemm_i8_kernel(
    const int8_t* __restrict__ Aq, const int8_t* __restrict__ Bq,
    const float* __restrict__ bias, const float* __restrict__ loraT,
    const float* __restrict__ loraB, const unsigned* __restrict__ slots,
    int slotA, int slotB, float* __restrict__ Out, int M, int N, int K,
    unsigned* absmaxSlot) {
    __shared__ __align__(16) char lA[2][128 * TPITCH];
    __shared__ __align__(16) char lB[2][128 * TPITCH];

    float sA = fmaxf(__uint_as_float(slots[slotA]) / 127.0f, 1e-8f);
    float sB = fmaxf(__uint_as_float(slots[slotB]) / 127.0f, 1e-8f);
    float sprod = sA * sB;

    int tid  = threadIdx.x;
    int wave = tid >> 5;
    int lane = tid & 31;
    int hi   = lane >> 4;
    int l16  = lane & 15;
    int waveM = wave & 3;
    int waveN = wave >> 2;

    int blockM = blockIdx.y * 128;
    int blockN = blockIdx.x * 128;

    auto issue_tiles = [&](int kt, int buf) {
#pragma unroll
        for (int j = 0; j < 4; ++j) {
            int i   = tid + j * 256;
            int row = i >> 3;
            int seg = (i & 7) * 16;
            async_copy_b128((unsigned)(uintptr_t)&lA[buf][row * TPITCH + seg],
                            Aq + (size_t)(blockM + row) * K + kt + seg);
            async_copy_b128((unsigned)(uintptr_t)&lB[buf][row * TPITCH + seg],
                            Bq + (size_t)(blockN + row) * K + kt + seg);
        }
    };

    v8i acc[2][4];
#pragma unroll
    for (int tm = 0; tm < 2; ++tm)
#pragma unroll
        for (int tn = 0; tn < 4; ++tn) acc[tm][tn] = (v8i)0;

    issue_tiles(0, 0);
    int buf = 0;
    for (int kt = 0; kt < K; kt += KSLAB) {
        wait_async0();
        __syncthreads();
        if (kt + KSLAB < K) issue_tiles(kt + KSLAB, buf ^ 1);

#pragma unroll
        for (int ks = 0; ks < KSLAB; ks += 64) {
            v8i a[2];
#pragma unroll
            for (int tm = 0; tm < 2; ++tm) {
                const char* ap =
                    &lA[buf][(waveM * 32 + tm * 16 + l16) * TPITCH + ks + hi * 8];
                int2 d0 = *(const int2*)(ap);
                int2 d1 = *(const int2*)(ap + 16);
                int2 d2 = *(const int2*)(ap + 32);
                int2 d3 = *(const int2*)(ap + 48);
                a[tm] = (v8i){d0.x, d0.y, d1.x, d1.y, d2.x, d2.y, d3.x, d3.y};
            }
            v8i b[4];
#pragma unroll
            for (int tn = 0; tn < 4; ++tn) {
                const char* bp =
                    &lB[buf][(waveN * 64 + tn * 16 + l16) * TPITCH + ks + hi * 8];
                int2 q0 = *(const int2*)(bp);
                int2 q1 = *(const int2*)(bp + 16);
                int2 q2 = *(const int2*)(bp + 32);
                int2 q3 = *(const int2*)(bp + 48);
                b[tn] = (v8i){q0.x, q0.y, q1.x, q1.y, q2.x, q2.y, q3.x, q3.y};
            }
#pragma unroll
            for (int tm = 0; tm < 2; ++tm)
#pragma unroll
                for (int tn = 0; tn < 4; ++tn)
                    acc[tm][tn] = __builtin_amdgcn_wmma_i32_16x16x64_iu8(
                        true, a[tm], true, b[tn], acc[tm][tn], false, false);
        }

        __syncthreads();
        buf ^= 1;
    }

    float4 lb0[4], lb1[4];
    float bv[4];
    int cols[4];
#pragma unroll
    for (int tn = 0; tn < 4; ++tn) {
        cols[tn] = blockN + waveN * 64 + tn * 16 + l16;
        const float4* lbp = (const float4*)(loraB + (size_t)cols[tn] * 8);
        lb0[tn] = lbp[0];
        lb1[tn] = lbp[1];
        bv[tn]  = bias[cols[tn]];
    }

    float localMax = 0.0f;
#pragma unroll
    for (int tm = 0; tm < 2; ++tm) {
#pragma unroll
        for (int j = 0; j < 8; ++j) {
            int row = blockM + waveM * 32 + tm * 16 + hi * 8 + j;
            const float4* tp = (const float4*)(loraT + (size_t)row * 8);
            float4 t0 = tp[0], t4 = tp[1];
            float* orow = Out + (size_t)row * N;
#pragma unroll
            for (int tn = 0; tn < 4; ++tn) {
                float lora = t0.x * lb0[tn].x + t0.y * lb0[tn].y +
                             t0.z * lb0[tn].z + t0.w * lb0[tn].w +
                             t4.x * lb1[tn].x + t4.y * lb1[tn].y +
                             t4.z * lb1[tn].z + t4.w * lb1[tn].w;
                float val = sprod * (float)acc[tm][tn][j] + bv[tn] + 2.0f * lora;
                if (FUSE_GELU_ABSMAX) {
                    val = 0.5f * val * (1.0f + erff(val * 0.70710678118654752f));
                    localMax = fmaxf(localMax, fabsf(val));
                }
                orow[cols[tn]] = val;
            }
        }
    }

    if (FUSE_GELU_ABSMAX) {
        __shared__ float smax[256];
        smax[tid] = localMax;
        __syncthreads();
        for (int s = 128; s > 0; s >>= 1) {
            if (tid < s) smax[tid] = fmaxf(smax[tid], smax[tid + s]);
            __syncthreads();
        }
        if (tid == 0) atomicMax(absmaxSlot, __float_as_uint(smax[0]));
    }
}

extern "C" void kernel_launch(void* const* d_in, const int* in_sizes, int n_in,
                              void* d_out, int out_size, void* d_ws, size_t ws_size,
                              hipStream_t stream) {
    const float* x    = (const float*)d_in[0];
    const float* W_fc = (const float*)d_in[1];
    const float* b_fc = (const float*)d_in[2];
    const float* A_fc = (const float*)d_in[3];
    const float* B_fc = (const float*)d_in[4];
    const float* W_pr = (const float*)d_in[5];
    const float* b_pr = (const float*)d_in[6];
    const float* A_pr = (const float*)d_in[7];
    const float* B_pr = (const float*)d_in[8];
    float* out = (float*)d_out;

    char* ws = (char*)d_ws;
    unsigned* slots = (unsigned*)(ws + OFF_SCALES);
    int8_t* xq   = (int8_t*)(ws + OFF_XQ);
    int8_t* wfcq = (int8_t*)(ws + OFF_WFCQ);
    int8_t* wpq  = (int8_t*)(ws + OFF_WPQ);
    float*  h    = (float*)(ws + OFF_H);
    int8_t* hq   = (int8_t*)(ws + OFF_HQ);
    float*  t1   = (float*)(ws + OFF_T1);
    float*  t2   = (float*)(ws + OFF_T2);

    qatmlp_init_kernel<<<1, 64, 0, stream>>>(slots);

    qatmlp_absmax_kernel<<<4096, 256, 0, stream>>>(x,    (size_t)M_DIM * K1_DIM, slots + 0);
    qatmlp_absmax_kernel<<<1024, 256, 0, stream>>>(W_fc, (size_t)N1_DIM * K1_DIM, slots + 1);
    qatmlp_absmax_kernel<<<1024, 256, 0, stream>>>(W_pr, (size_t)N2_DIM * K2_DIM, slots + 2);

    qatmlp_quant_kernel<<<1024, 256, 0, stream>>>(W_fc, wfcq, (size_t)N1_DIM * K1_DIM / 4, slots + 1);
    qatmlp_quant_kernel<<<1024, 256, 0, stream>>>(W_pr, wpq,  (size_t)N2_DIM * K2_DIM / 4, slots + 2);

    qatmlp_rowquant_lora_kernel<<<M_DIM, 256, 0, stream>>>(x, A_fc, xq, t1, K1_DIM, slots + 0);

    qatmlp_gemm_i8_kernel<true><<<dim3(N1_DIM / 128, M_DIM / 128), 256, 0, stream>>>(
        xq, wfcq, b_fc, t1, B_fc, slots, 0, 1, h, M_DIM, N1_DIM, K1_DIM, slots + 3);

    qatmlp_rowquant_lora_kernel<<<M_DIM, 256, 0, stream>>>(h, A_pr, hq, t2, K2_DIM, slots + 3);

    qatmlp_gemm_i8_kernel<false><<<dim3(N2_DIM / 128, M_DIM / 128), 256, 0, stream>>>(
        hq, wpq, b_pr, t2, B_pr, slots, 3, 2, out, M_DIM, N2_DIM, K2_DIM, nullptr);
}
